// GNN_7834020348615
// MI455X (gfx1250) — matrix operand order repaired and checked by reading, not compiled, not run on hardware
//
#include <hip/hip_runtime.h>

// ---------------- problem constants (match reference) ----------------
#define GNN_N   50000
#define GNN_E   200000
#define GNN_H   512
#define GNN_H2  1024
#define GNN_L   4
#define GNN_AF  9
#define GNN_BF  3
#define GNN_AV  128
#define GNN_BV  16
#define GNN_BN_EPS 1e-5f

typedef __bf16 bf16;
typedef __attribute__((ext_vector_type(16))) __bf16 v16bf;
typedef __attribute__((ext_vector_type(4)))  __bf16 bf16x4;
typedef __attribute__((ext_vector_type(8)))  float  v8f;

union FragBits { v16bf v; uint4 q[2]; };

// =====================================================================
// Atom encoder: h[n,:] = sum_f atom_emb[f, x[n,f], :]
// =====================================================================
__global__ __launch_bounds__(128)
void gnn_atom_encode(const int* __restrict__ x, const float* __restrict__ atom_emb,
                     float* __restrict__ h)
{
    const int n = blockIdx.x;
    const int c = threadIdx.x * 4;
    float ax = 0.f, ay = 0.f, az = 0.f, aw = 0.f;
#pragma unroll
    for (int f = 0; f < GNN_AF; ++f) {
        const int v = x[n * GNN_AF + f];
        const float4 t = *(const float4*)&atom_emb[((size_t)(f * GNN_AV + v)) * GNN_H + c];
        ax += t.x; ay += t.y; az += t.z; aw += t.w;
    }
    float4 o; o.x = ax; o.y = ay; o.z = az; o.w = aw;
    *(float4*)&h[(size_t)n * GNN_H + c] = o;
}

// =====================================================================
// inter = (1 + eps[l]) * h   (grid-stride float4)
// =====================================================================
__global__ __launch_bounds__(256)
void gnn_init_inter(const float* __restrict__ h, const float* __restrict__ eps, int l,
                    float* __restrict__ inter)
{
    const float s = 1.0f + eps[l];
    const size_t total = (size_t)GNN_N * GNN_H / 4;
    for (size_t i = (size_t)blockIdx.x * blockDim.x + threadIdx.x; i < total;
         i += (size_t)gridDim.x * blockDim.x) {
        float4 v = ((const float4*)h)[i];
        v.x *= s; v.y *= s; v.z *= s; v.w *= s;
        ((float4*)inter)[i] = v;
    }
}

// =====================================================================
// Edge kernel: msg = relu(h[src] + bond_embed(e)); inter[dst] += msg
// =====================================================================
__global__ __launch_bounds__(128)
void gnn_edge_msg(const float* __restrict__ h, const int* __restrict__ ei,
                  const int* __restrict__ ea, const float* __restrict__ bemb,
                  float* __restrict__ inter)
{
    const int e = blockIdx.x;
    const int src = ei[e];
    const int dst = ei[GNN_E + e];
    const int a0 = ea[e * GNN_BF + 0];
    const int a1 = ea[e * GNN_BF + 1];
    const int a2 = ea[e * GNN_BF + 2];
    const float* t0 = bemb + ((size_t)(0 * GNN_BV + a0)) * GNN_H;
    const float* t1 = bemb + ((size_t)(1 * GNN_BV + a1)) * GNN_H;
    const float* t2 = bemb + ((size_t)(2 * GNN_BV + a2)) * GNN_H;
    const float* hs = h + (size_t)src * GNN_H;
    float* id = inter + (size_t)dst * GNN_H;
    const int c = threadIdx.x * 4;

    const float4 v0 = *(const float4*)(t0 + c);
    const float4 v1 = *(const float4*)(t1 + c);
    const float4 v2 = *(const float4*)(t2 + c);
    const float4 vh = *(const float4*)(hs + c);
    atomicAdd(&id[c + 0], fmaxf(vh.x + v0.x + v1.x + v2.x, 0.f));
    atomicAdd(&id[c + 1], fmaxf(vh.y + v0.y + v1.y + v2.y, 0.f));
    atomicAdd(&id[c + 2], fmaxf(vh.z + v0.z + v1.z + v2.z, 0.f));
    atomicAdd(&id[c + 3], fmaxf(vh.w + v0.w + v1.w + v2.w, 0.f));
}

// =====================================================================
// WMMA GEMM: C[M x Nc] = A[M x K] @ B[K x Nc] + bias
//
// fp32 inputs converted to bf16 while staging into LDS;
// v_wmma_f32_16x16x32_bf16 with fp32 accumulation.
//
// Block: 128 threads = 4 wave32. Block tile 128 rows x 64 cols; wave w
// owns rows [w*32, w*32+32) x all 4 column tiles -> 8 accumulators.
// Software pipeline: double-buffered LDS; next K-tile's global loads are
// issued before computing the current tile (one barrier per K-step).
//
// Fragment layouts per CDNA5 ISA 7.12.2:
//  A (16x32): m = lane&15, half = lane>>4;
//    elems 0..7  -> K = 8*half + i ; elems 8..15 -> K = 16 + 8*half + (i-8)
//  B (32x16): lane holds column n = lane&15; K mapping same as A
//    (column-major B staged transposed in LDS so K is contiguous)
// =====================================================================

// per-thread staging register counts for one K-step (K-tile = 32)
//   A: 128 rows x 8 float4  = 1024 quads / 128 thr = 8
//   B:  32 k   x 16 float4  =  512 quads / 128 thr = 4

__device__ __forceinline__ void gemm_gload_A(const float* __restrict__ A, int rowBase,
                                             int M, int K, int k0, int tid, float4 ra[8])
{
#pragma unroll
    for (int i = 0; i < 8; ++i) {
        const int idx = tid + i * 128;
        const int r   = idx >> 3;
        const int c4  = (idx & 7) * 4;
        const int gr  = rowBase + r;
        const int grc = gr < M ? gr : (M - 1);          // clamp: load always valid
        float4 v = *(const float4*)&A[(size_t)grc * K + k0 + c4];
        if (gr >= M) { v.x = 0.f; v.y = 0.f; v.z = 0.f; v.w = 0.f; }  // branch-free select
        ra[i] = v;
    }
}

__device__ __forceinline__ void gemm_gload_B(const float* __restrict__ B, int colBase,
                                             int Nc, int k0, int tid, float4 rb[4])
{
#pragma unroll
    for (int i = 0; i < 4; ++i) {
        const int idx = tid + i * 128;
        const int kk  = idx >> 4;
        const int c4  = (idx & 15) * 4;
        rb[i] = *(const float4*)&B[(size_t)(k0 + kk) * Nc + colBase + c4];
    }
}

__device__ __forceinline__ void gemm_stage_A(bf16 (*As)[40], int tid, const float4 ra[8])
{
#pragma unroll
    for (int i = 0; i < 8; ++i) {
        const int idx = tid + i * 128;
        const int r   = idx >> 3;
        const int c4  = (idx & 7) * 4;
        bf16x4 b;
        b[0] = (bf16)ra[i].x; b[1] = (bf16)ra[i].y;
        b[2] = (bf16)ra[i].z; b[3] = (bf16)ra[i].w;
        *(bf16x4*)&As[r][c4] = b;
    }
}

__device__ __forceinline__ void gemm_stage_B(bf16 (*Bt)[40], int tid, const float4 rb[4])
{
#pragma unroll
    for (int i = 0; i < 4; ++i) {
        const int idx = tid + i * 128;
        const int kk  = idx >> 4;
        const int c4  = (idx & 15) * 4;
        Bt[c4 + 0][kk] = (bf16)rb[i].x;
        Bt[c4 + 1][kk] = (bf16)rb[i].y;
        Bt[c4 + 2][kk] = (bf16)rb[i].z;
        Bt[c4 + 3][kk] = (bf16)rb[i].w;
    }
}

__global__ __launch_bounds__(128)
void gnn_gemm_wmma(const float* __restrict__ A, const float* __restrict__ B,
                   const float* __restrict__ bias, float* __restrict__ C,
                   int M, int K, int Nc)
{
    __shared__ bf16 As[2][128][40];   // 2 x 10240 B
    __shared__ bf16 Bt[2][64][40];    // 2 x  5120 B  (transposed: [col][k])

    const int tid  = threadIdx.x;
    const int wave = tid >> 5;
    const int lane = tid & 31;
    const int hl   = lane >> 4;
    const int ln   = lane & 15;
    const int rowBase = blockIdx.x * 128;
    const int colBase = blockIdx.y * 64;

    v8f acc[2][4] = {};
    float4 ra[8], rb[4];

    // ---- prologue: tile 0 -> LDS buffer 0
    gemm_gload_A(A, rowBase, M, K, 0, tid, ra);
    gemm_gload_B(B, colBase, Nc, 0, tid, rb);
    gemm_stage_A(As[0], tid, ra);
    gemm_stage_B(Bt[0], tid, rb);
    __syncthreads();

    const int nk = K >> 5;
    for (int ks = 0; ks < nk; ++ks) {
        const int cur = ks & 1;

        // issue next tile's global loads before computing (latency hiding)
        if (ks + 1 < nk) {
            gemm_gload_A(A, rowBase, M, K, (ks + 1) * 32, tid, ra);
            gemm_gload_B(B, colBase, Nc, (ks + 1) * 32, tid, rb);
        }

        // ---- compute from LDS buffer `cur`
        FragBits fa0, fa1;
        {
            const int r0 = wave * 32 + ln;
            fa0.q[0] = *(const uint4*)&As[cur][r0][8 * hl];
            fa0.q[1] = *(const uint4*)&As[cur][r0][16 + 8 * hl];
            fa1.q[0] = *(const uint4*)&As[cur][r0 + 16][8 * hl];
            fa1.q[1] = *(const uint4*)&As[cur][r0 + 16][16 + 8 * hl];
        }
#pragma unroll
        for (int tn = 0; tn < 4; ++tn) {
            FragBits fb;
            const int c = tn * 16 + ln;
            fb.q[0] = *(const uint4*)&Bt[cur][c][8 * hl];
            fb.q[1] = *(const uint4*)&Bt[cur][c][16 + 8 * hl];
            acc[0][tn] = __builtin_amdgcn_wmma_f32_16x16x32_bf16(
                false, fa0.v, false, fb.v, (short)0, acc[0][tn], false, false);
            acc[1][tn] = __builtin_amdgcn_wmma_f32_16x16x32_bf16(
                false, fa1.v, false, fb.v, (short)0, acc[1][tn], false, false);
        }

        // ---- stage next tile into the other buffer; one barrier per step
        if (ks + 1 < nk) {
            gemm_stage_A(As[cur ^ 1], tid, ra);
            gemm_stage_B(Bt[cur ^ 1], tid, rb);
            __syncthreads();
        }
    }

    // ---- epilogue: C layout (VGPR v: M = v + 8*hl, N = lane&15), add bias
#pragma unroll
    for (int tn = 0; tn < 4; ++tn) {
        const int col = colBase + tn * 16 + ln;
        const float bs = bias[col];
#pragma unroll
        for (int tm = 0; tm < 2; ++tm) {
#pragma unroll
            for (int v = 0; v < 8; ++v) {
                const int row = rowBase + wave * 32 + tm * 16 + v + 8 * hl;
                if (row < M) C[(size_t)row * Nc + col] = acc[tm][tn][v] + bs;
            }
        }
    }
}

// =====================================================================
// BN helpers
// =====================================================================
__global__ __launch_bounds__(256)
void gnn_zero(float* __restrict__ p, int n)
{
    const int i = blockIdx.x * blockDim.x + threadIdx.x;
    if (i < n) p[i] = 0.f;
}

// Column partial sums / sum-of-squares. grid = (C/256, rowChunks), block 256.
__global__ __launch_bounds__(256)
void gnn_bn_stats(const float* __restrict__ z, int M, int C, float* __restrict__ stats)
{
    const int col = blockIdx.x * 256 + threadIdx.x;
    const int r0  = blockIdx.y * 256;
    const int r1  = min(r0 + 256, M);
    float s = 0.f, sq = 0.f;
    for (int r = r0; r < r1; ++r) {
        const float v = z[(size_t)r * C + col];
        s  += v;
        sq += v * v;
    }
    atomicAdd(&stats[col], s);
    atomicAdd(&stats[C + col], sq);
}

// y = (z - mean) * rsqrt(var + eps) * g + b ; optional relu; out may == z
__global__ __launch_bounds__(256)
void gnn_bn_apply(const float* __restrict__ z, const float* __restrict__ stats,
                  const float* __restrict__ g, const float* __restrict__ b,
                  float* __restrict__ out, int M, int C, int do_relu)
{
    const float invM = 1.0f / (float)M;
    const size_t total = (size_t)M * C / 4;
    const int cq = C / 4;
    for (size_t i = (size_t)blockIdx.x * blockDim.x + threadIdx.x; i < total;
         i += (size_t)gridDim.x * blockDim.x) {
        const int c = (int)(i % cq) * 4;
        float4 v = ((const float4*)z)[i];
        float* vp = &v.x;
#pragma unroll
        for (int j = 0; j < 4; ++j) {
            const int cc = c + j;
            const float mean = stats[cc] * invM;
            const float var  = stats[C + cc] * invM - mean * mean;
            const float sc   = rsqrtf(var + GNN_BN_EPS) * g[cc];
            float y = (vp[j] - mean) * sc + b[cc];
            if (do_relu) y = fmaxf(y, 0.f);
            vp[j] = y;
        }
        ((float4*)out)[i] = v;
    }
}

// =====================================================================
// Host-side orchestration
// =====================================================================
extern "C" void kernel_launch(void* const* d_in, const int* in_sizes, int n_in,
                              void* d_out, int out_size, void* d_ws, size_t ws_size,
                              hipStream_t stream)
{
    (void)in_sizes; (void)n_in; (void)out_size; (void)ws_size;

    const int*   x         = (const int*)  d_in[0];
    const int*   edge_idx  = (const int*)  d_in[1];
    const int*   edge_attr = (const int*)  d_in[2];
    const float* atom_emb  = (const float*)d_in[3];
    const float* bond_emb  = (const float*)d_in[4];
    const float* eps       = (const float*)d_in[5];
    const float* W1        = (const float*)d_in[6];
    const float* b1        = (const float*)d_in[7];
    const float* g1        = (const float*)d_in[8];
    const float* be1       = (const float*)d_in[9];
    const float* W2        = (const float*)d_in[10];
    const float* b2        = (const float*)d_in[11];
    const float* bn_g      = (const float*)d_in[12];
    const float* bn_b      = (const float*)d_in[13];
    float*       out       = (float*)d_out;

    // workspace layout (floats)
    const size_t NH  = (size_t)GNN_N * GNN_H;    // 25.6M
    const size_t NH2 = (size_t)GNN_N * GNN_H2;   // 51.2M
    float* hA    = (float*)d_ws;
    float* hB    = hA + NH;
    float* inter = hB + NH;       // also reused as z2 buffer
    float* z1    = inter + NH;
    float* stats = z1 + NH2;      // 2*H2 floats used max

    const int ewGrid = 4096;
    const dim3 gemmBlk(128);
    const dim3 g1Grid((GNN_N + 127) / 128, GNN_H2 / 64);  // 391 x 16
    const dim3 g2Grid((GNN_N + 127) / 128, GNN_H  / 64);  // 391 x 8
    const dim3 stat1Grid(GNN_H2 / 256, (GNN_N + 255) / 256);
    const dim3 stat2Grid(GNN_H  / 256, (GNN_N + 255) / 256);

    gnn_atom_encode<<<GNN_N, 128, 0, stream>>>(x, atom_emb, hA);

    float* h_cur = hA;
    float* h_nxt = hB;
    for (int l = 0; l < GNN_L; ++l) {
        const float* bemb_l = bond_emb + (size_t)l * GNN_BF * GNN_BV * GNN_H;

        gnn_init_inter<<<ewGrid, 256, 0, stream>>>(h_cur, eps, l, inter);
        gnn_edge_msg<<<GNN_E, 128, 0, stream>>>(h_cur, edge_idx, edge_attr, bemb_l, inter);

        gnn_gemm_wmma<<<g1Grid, gemmBlk, 0, stream>>>(
            inter, W1 + (size_t)l * GNN_H * GNN_H2, b1 + (size_t)l * GNN_H2,
            z1, GNN_N, GNN_H, GNN_H2);
        gnn_zero<<<(2 * GNN_H2 + 255) / 256, 256, 0, stream>>>(stats, 2 * GNN_H2);
        gnn_bn_stats<<<stat1Grid, 256, 0, stream>>>(z1, GNN_N, GNN_H2, stats);
        gnn_bn_apply<<<ewGrid, 256, 0, stream>>>(
            z1, stats, g1 + (size_t)l * GNN_H2, be1 + (size_t)l * GNN_H2,
            z1, GNN_N, GNN_H2, 1);

        gnn_gemm_wmma<<<g2Grid, gemmBlk, 0, stream>>>(
            z1, W2 + (size_t)l * GNN_H2 * GNN_H, b2 + (size_t)l * GNN_H,
            inter, GNN_N, GNN_H2, GNN_H);
        gnn_zero<<<(2 * GNN_H + 255) / 256, 256, 0, stream>>>(stats, 2 * GNN_H);
        gnn_bn_stats<<<stat2Grid, 256, 0, stream>>>(inter, GNN_N, GNN_H, stats);

        float* dst = (l == GNN_L - 1) ? out : h_nxt;
        gnn_bn_apply<<<ewGrid, 256, 0, stream>>>(
            inter, stats, bn_g + (size_t)l * GNN_H, bn_b + (size_t)l * GNN_H,
            dst, GNN_N, GNN_H, (l != GNN_L - 1) ? 1 : 0);

        float* t = h_cur; h_cur = h_nxt; h_nxt = t;
    }
}
